// LongformerMLM_42279658062073
// MI455X (gfx1250) — matrix operand order repaired and checked by reading, not compiled, not run on hardware
//
#include <hip/hip_runtime.h>
#include <hip/hip_bf16.h>

typedef __attribute__((ext_vector_type(16))) _Float16 v16h;
typedef __attribute__((ext_vector_type(8)))  _Float16 half8;
typedef __attribute__((ext_vector_type(8)))  float    v8f;
typedef int v4i __attribute__((vector_size(16)));

#define S_   4096
#define D_   512
#define H_   8
#define HD_  64
#define G_   8
#define FF_  2048
#define V_   8192
#define WW_  256        // w = WIN/2
#define NC_  16
#define KW_  768        // 3*w
#define GPAD 32         // global slot padded 8 -> 32
#define PCOLS (GPAD + KW_)   // 800
#define NEGF (-60000.0f)     // finite in f16; exp() underflows to 0

#ifndef __has_builtin
#define __has_builtin(x) 0
#endif
#if __has_builtin(__builtin_amdgcn_global_load_async_to_lds_b128)
#define ASYNC_LDS 1
#else
#define ASYNC_LDS 0
#endif

// ---------------- async global -> LDS helpers (CDNA5 ASYNCcnt path) ---------
__device__ inline void async_cp16(const _Float16* g, _Float16* l) {
#if ASYNC_LDS
    __builtin_amdgcn_global_load_async_to_lds_b128(
        (__attribute__((address_space(1))) v4i*)g,
        (__attribute__((address_space(3))) v4i*)l, 0, 0);
#else
    *(half8*)l = *(const half8*)g;   // synchronous fallback
#endif
}
__device__ inline void async_wait0() {
#if ASYNC_LDS
#if __has_builtin(__builtin_amdgcn_s_wait_asynccnt)
    __builtin_amdgcn_s_wait_asynccnt(0);
#else
    asm volatile("s_wait_asynccnt 0x0" ::: "memory");
#endif
#endif
}
__device__ inline void async_wait2() {
#if ASYNC_LDS
#if __has_builtin(__builtin_amdgcn_s_wait_asynccnt)
    __builtin_amdgcn_s_wait_asynccnt(2);
#else
    asm volatile("s_wait_asynccnt 0x2" ::: "memory");
#endif
#endif
}

// ---------------- WMMA fragment helpers -------------------------------------
// A (16x32 f16): lane L -> row m = L&15, k = 16*(L>>4)+{0..15}
__device__ inline v16h load_a_frag(const _Float16* A, int lda, int row0, int k0, int lane) {
    int m  = lane & 15;
    int kb = k0 + ((lane >> 4) << 4);
    const _Float16* p = A + (size_t)(row0 + m) * lda + kb;
    half8 lo = *(const half8*)(p);
    half8 hi = *(const half8*)(p + 8);
    return __builtin_shufflevector(lo, hi, 0,1,2,3,4,5,6,7,8,9,10,11,12,13,14,15);
}
// B (32x16 f16) from Bt laid out [N][K]: lane L -> col n=L&15, k = 16*(L>>4)+{0..15}
__device__ inline v16h load_b_frag(const _Float16* Bt, size_t ldb, int n0, int k0, int lane) {
    int n  = lane & 15;
    int kb = k0 + ((lane >> 4) << 4);
    return *(const v16h*)(Bt + (size_t)(n0 + n) * ldb + kb);
}
// B fragment from an LDS tile laid out [64 rows][32 halves]
__device__ inline v16h load_b_frag_lds(const _Float16* sB, int nrel0, int lane) {
    int n  = lane & 15;
    int kb = (lane >> 4) << 4;
    return *(const v16h*)(sB + (size_t)(nrel0 + n) * 32 + kb);
}

// ---------------- generic WMMA GEMM (4 waves, LDS-staged B, double-buffer) --
// C[M,N] = act((A16[M,K] @ Bt16[N,K]^T + bias) * scale) (+ resid); writes f32/f16
__global__ __launch_bounds__(128) void k_gemm(
    const _Float16* __restrict__ A, const _Float16* __restrict__ Bt,
    const float* __restrict__ bias, const float* __restrict__ resid,
    float* __restrict__ C32, _Float16* __restrict__ C16,
    int M, int N, int K, float scale, int act)
{
    __shared__ _Float16 sB[2][64 * 32];   // two 64(N) x 32(K) tiles, 4KB each
    int t    = threadIdx.x;
    int lane = t & 31;
    int wid  = t >> 5;
    int n0 = blockIdx.x * 64;
    int m0 = blockIdx.y * 128 + wid * 32;

    // cooperative copy slice: each of 128 threads moves 32B (two B128 ops)
    int crow = t >> 1;            // 0..63 : relative N row
    int cofs = (t & 1) * 16;      // 0 or 16 halves within the 32-half row

#define CPTILE(buf, kk) do {                                                   \
        const _Float16* g = Bt + (size_t)(n0 + crow) * K + (kk) + cofs;        \
        _Float16* l = &sB[buf][crow * 32 + cofs];                              \
        async_cp16(g, l); async_cp16(g + 8, l + 8);                            \
    } while (0)

    v8f zero = {};
    v8f acc[2][4];
    for (int i = 0; i < 2; ++i) for (int j = 0; j < 4; ++j) acc[i][j] = zero;

    int nk = K >> 5;
    CPTILE(0, 0);
    for (int ki = 0; ki < nk; ++ki) {
        int buf = ki & 1;
        if (ki + 1 < nk) { CPTILE(buf ^ 1, (ki + 1) * 32); async_wait2(); }
        else             { async_wait0(); }
        __syncthreads();

        int k0 = ki * 32;
        __builtin_prefetch((const void*)(A + (size_t)(m0 + (lane & 15)) * K + k0 + 64), 0, 1);
        v16h a0 = load_a_frag(A, K, m0,      k0, lane);
        v16h a1 = load_a_frag(A, K, m0 + 16, k0, lane);
#pragma unroll
        for (int j = 0; j < 4; ++j) {
            v16h b = load_b_frag_lds(&sB[buf][0], j * 16, lane);
            acc[0][j] = __builtin_amdgcn_wmma_f32_16x16x32_f16(false, a0, false, b, (short)0, acc[0][j], false, false);
            acc[1][j] = __builtin_amdgcn_wmma_f32_16x16x32_f16(false, a1, false, b, (short)0, acc[1][j], false, false);
        }
        __syncthreads();   // buf may be overwritten next iteration
    }
#undef CPTILE

    int halfsel = lane >> 4, col0 = lane & 15;
    for (int i = 0; i < 2; ++i) {
        for (int j = 0; j < 4; ++j) {
            int col = n0 + j * 16 + col0;
            float bv = bias ? bias[col] : 0.0f;
            for (int r = 0; r < 8; ++r) {
                int row = m0 + i * 16 + halfsel * 8 + r;
                float v = (acc[i][j][r] + bv) * scale;
                if (act == 1) v = 0.5f * v * (1.0f + erff(v * 0.70710678118f));
                if (resid) v += resid[(size_t)row * N + col];
                if (C32) C32[(size_t)row * N + col] = v;
                if (C16) C16[(size_t)row * N + col] = (_Float16)v;
            }
        }
    }
}

// ---------------- fused local (windowed) attention --------------------------
// one wave per (head, chunk, 16-query block): scores -> softmax -> PV, all WMMA
__global__ __launch_bounds__(32) void k_local_attn(
    const _Float16* __restrict__ q16, const _Float16* __restrict__ k16,
    const _Float16* __restrict__ vT16, _Float16* __restrict__ out16,
    const int* __restrict__ amask, const int* __restrict__ gmask)
{
    __shared__ _Float16 sP[16][PCOLS];
    int lane = threadIdx.x;
    int rb = blockIdx.x;          // 16-row block within chunk (0..15)
    int c  = blockIdx.y;          // chunk (0..15)
    int h  = blockIdx.z;          // head
    int srow0 = c * WW_ + rb * 16;
    int colh  = h * HD_;
    int halfsel = lane >> 4, nn = lane & 15;
    v8f zero = {};

    // Q fragments are invariant over all 50 score tiles: load once
    v16h aq0 = load_a_frag(q16, D_, srow0, colh + 0,  lane);
    v16h aq1 = load_a_frag(q16, D_, srow0, colh + 32, lane);

    // global score tiles: keys 0..31 (8 real, rest -> NEG)
    for (int gt = 0; gt < 2; ++gt) {
        v8f acc = zero;
        int key = gt * 16 + nn;
        v16h b0 = *(const v16h*)(k16 + (size_t)key * D_ + colh + 0  + halfsel * 16);
        v16h b1 = *(const v16h*)(k16 + (size_t)key * D_ + colh + 32 + halfsel * 16);
        acc = __builtin_amdgcn_wmma_f32_16x16x32_f16(false, aq0, false, b0, (short)0, acc, false, false);
        acc = __builtin_amdgcn_wmma_f32_16x16x32_f16(false, aq1, false, b1, (short)0, acc, false, false);
        for (int r = 0; r < 8; ++r)
            sP[halfsel * 8 + r][gt * 16 + nn] = (_Float16)((key < G_) ? acc[r] : NEGF);
    }
    // local score tiles: 48 x 16 keys around the chunk
    for (int jt = 0; jt < 48; ++jt) {
        v8f acc = zero;
        int j_n = jt * 16 + nn;                 // 0..767
        int key = c * WW_ + j_n - WW_;          // key_abs
        int kc  = key < 0 ? 0 : (key >= S_ ? S_ - 1 : key);
        v16h b0 = *(const v16h*)(k16 + (size_t)kc * D_ + colh + 0  + halfsel * 16);
        v16h b1 = *(const v16h*)(k16 + (size_t)kc * D_ + colh + 32 + halfsel * 16);
        acc = __builtin_amdgcn_wmma_f32_16x16x32_f16(false, aq0, false, b0, (short)0, acc, false, false);
        acc = __builtin_amdgcn_wmma_f32_16x16x32_f16(false, aq1, false, b1, (short)0, acc, false, false);
        for (int r = 0; r < 8; ++r) {
            int rowc = rb * 16 + halfsel * 8 + r;   // query row within chunk
            int rel  = j_n - WW_ - rowc;
            bool ok = (rel >= -WW_) && (rel <= WW_) && (key >= 0) && (key < S_);
            if (ok) ok = (amask[key] > 0) && (gmask[key] == 0);
            sP[halfsel * 8 + r][GPAD + j_n] = (_Float16)(ok ? acc[r] : NEGF);
        }
    }
    __syncthreads();
    // per-row softmax over 800 columns
    if (lane < 16) {
        float mx = -3.0e38f;
        for (int jj = 0; jj < PCOLS; ++jj) { float v = (float)sP[lane][jj]; mx = v > mx ? v : mx; }
        float sum = 0.0f;
        for (int jj = 0; jj < PCOLS; ++jj) { float e = __expf((float)sP[lane][jj] - mx); sum += e; sP[lane][jj] = (_Float16)e; }
        float inv = 1.0f / sum;
        for (int jj = 0; jj < PCOLS; ++jj) sP[lane][jj] = (_Float16)((float)sP[lane][jj] * inv);
    }
    __syncthreads();
    // PV: out(16x64) = P(16x800) @ Vcat(800x64), B from vT16[d][s]
    v8f oacc[4];
    for (int j = 0; j < 4; ++j) oacc[j] = zero;
    for (int k0 = 0; k0 < PCOLS; k0 += 32) {
        v16h a = load_a_frag(&sP[0][0], PCOLS, 0, k0, lane);
        int keyb = (k0 < GPAD) ? (k0 + halfsel * 16)
                               : (c * WW_ - WW_ + (k0 - GPAD) + halfsel * 16);
        int kb = keyb < 0 ? 0 : (keyb > S_ - 16 ? S_ - 16 : keyb);  // P==0 where clamped
#pragma unroll
        for (int j = 0; j < 4; ++j) {
            v16h b = *(const v16h*)(vT16 + (size_t)(colh + j * 16 + nn) * S_ + kb);
            oacc[j] = __builtin_amdgcn_wmma_f32_16x16x32_f16(false, a, false, b, (short)0, oacc[j], false, false);
        }
    }
    for (int j = 0; j < 4; ++j)
        for (int r = 0; r < 8; ++r) {
            int row = srow0 + halfsel * 8 + r;
            out16[(size_t)row * D_ + colh + j * 16 + nn] = (_Float16)oacc[j][r];
        }
}

// ---------------- global attention (8 query rows, full S keys) --------------
__global__ __launch_bounds__(128) void k_global_attn(
    const _Float16* __restrict__ qg, const _Float16* __restrict__ kg,
    const _Float16* __restrict__ vg, _Float16* __restrict__ out16,
    const int* __restrict__ amask)
{
    __shared__ float scr[S_];
    __shared__ float red[128];
    int t = threadIdx.x;
    int h = blockIdx.x >> 3;
    int g = blockIdx.x & 7;
    const _Float16* qrow = qg + (size_t)g * D_ + h * HD_;
    for (int s = t; s < S_; s += 128) {
        float acc = 0.0f;
        const _Float16* krow = kg + (size_t)s * D_ + h * HD_;
        for (int d = 0; d < HD_; ++d) acc += (float)qrow[d] * (float)krow[d];
        scr[s] = (amask[s] > 0) ? acc : -1.0e9f;
    }
    __syncthreads();
    float mx = -3.0e38f;
    for (int s = t; s < S_; s += 128) mx = fmaxf(mx, scr[s]);
    red[t] = mx; __syncthreads();
    for (int off = 64; off > 0; off >>= 1) { if (t < off) red[t] = fmaxf(red[t], red[t + off]); __syncthreads(); }
    mx = red[0]; __syncthreads();
    float sum = 0.0f;
    for (int s = t; s < S_; s += 128) { float e = __expf(scr[s] - mx); scr[s] = e; sum += e; }
    red[t] = sum; __syncthreads();
    for (int off = 64; off > 0; off >>= 1) { if (t < off) red[t] += red[t + off]; __syncthreads(); }
    float inv = 1.0f / red[0];
    __syncthreads();
    if (t < HD_) {
        float acc = 0.0f;
        for (int s = 0; s < S_; ++s) acc += scr[s] * (float)vg[(size_t)s * D_ + h * HD_ + t];
        out16[(size_t)g * D_ + h * HD_ + t] = (_Float16)(acc * inv);
    }
}

// ---------------- LayerNorm (row-wise over D=512), writes f32 + f16 ---------
__global__ __launch_bounds__(128) void k_layernorm(
    const float* __restrict__ x, const float* __restrict__ gw, const float* __restrict__ bw,
    float* __restrict__ y32, _Float16* __restrict__ y16)
{
    __shared__ float red[128];
    int row = blockIdx.x, t = threadIdx.x;
    const float* xr = x + (size_t)row * D_;
    float s = 0.0f;
    for (int d = t; d < D_; d += 128) s += xr[d];
    red[t] = s; __syncthreads();
    for (int off = 64; off > 0; off >>= 1) { if (t < off) red[t] += red[t + off]; __syncthreads(); }
    float mean = red[0] * (1.0f / D_); __syncthreads();
    float v = 0.0f;
    for (int d = t; d < D_; d += 128) { float dd = xr[d] - mean; v += dd * dd; }
    red[t] = v; __syncthreads();
    for (int off = 64; off > 0; off >>= 1) { if (t < off) red[t] += red[t + off]; __syncthreads(); }
    float rstd = rsqrtf(red[0] * (1.0f / D_) + 1e-12f);
    for (int d = t; d < D_; d += 128) {
        float o = (xr[d] - mean) * rstd * gw[d] + bw[d];
        y32[(size_t)row * D_ + d] = o;
        y16[(size_t)row * D_ + d] = (_Float16)o;
    }
}

// ---------------- small elementwise kernels ---------------------------------
__global__ void k_embed(const int* __restrict__ x, const float* __restrict__ we,
                        const float* __restrict__ pe, const float* __restrict__ te,
                        float* __restrict__ out)
{
    int i = blockIdx.x * blockDim.x + threadIdx.x;
    if (i >= S_ * D_) return;
    int s = i / D_, d = i - s * D_;
    out[i] = we[(size_t)x[s] * D_ + d] + pe[(size_t)(s + 2) * D_ + d] + te[d];
}

__global__ void k_cvtT(const float* __restrict__ W, _Float16* __restrict__ Wt, int K, int N)
{
    long long i = (long long)blockIdx.x * blockDim.x + threadIdx.x;
    if (i >= (long long)K * N) return;
    int k = (int)(i / N), n = (int)(i - (long long)k * N);
    Wt[(size_t)n * K + k] = (_Float16)W[i];
}

__global__ void k_vtrans(const _Float16* __restrict__ v, _Float16* __restrict__ vT)
{
    int i = blockIdx.x * blockDim.x + threadIdx.x;
    if (i >= S_ * D_) return;
    int s = i / D_, c = i - s * D_;
    vT[(size_t)c * S_ + s] = v[i];
}

__global__ void k_copyf(const float* __restrict__ src, float* __restrict__ dst, int n)
{
    int i = blockIdx.x * blockDim.x + threadIdx.x;
    if (i < n) dst[i] = src[i];
}

// ---------------- host driver -----------------------------------------------
extern "C" void kernel_launch(void* const* d_in, const int* in_sizes, int n_in,
                              void* d_out, int out_size, void* d_ws, size_t ws_size,
                              hipStream_t stream)
{
    (void)in_sizes; (void)n_in; (void)out_size; (void)ws_size;
    const int*   x      = (const int*)d_in[0];
    const int*   amask  = (const int*)d_in[1];
    const int*   gmask  = (const int*)d_in[2];
    const float* we     = (const float*)d_in[3];
    const float* pe     = (const float*)d_in[4];
    const float* te     = (const float*)d_in[5];
    const float* elng   = (const float*)d_in[6];
    const float* elnb   = (const float*)d_in[7];
    const float* W_out  = (const float*)d_in[96];
    const float* b_out  = (const float*)d_in[97];
    float* out = (float*)d_out;

    size_t off = 0;
    auto alloc = [&](size_t bytes) {
        void* p = (char*)d_ws + off;
        off += (bytes + 255) & ~(size_t)255;
        return p;
    };
    float*    tmp32  = (float*)   alloc((size_t)S_ * D_ * 4);
    float*    h32    = (float*)   alloc((size_t)S_ * D_ * 4);
    _Float16* h16    = (_Float16*)alloc((size_t)S_ * D_ * 2);
    _Float16* q16    = (_Float16*)alloc((size_t)S_ * D_ * 2);
    _Float16* k16    = (_Float16*)alloc((size_t)S_ * D_ * 2);
    _Float16* v16    = (_Float16*)alloc((size_t)S_ * D_ * 2);
    _Float16* vT16   = (_Float16*)alloc((size_t)S_ * D_ * 2);
    _Float16* attn16 = (_Float16*)alloc((size_t)S_ * D_ * 2);
    _Float16* qg16   = (_Float16*)alloc((size_t)S_ * D_ * 2);
    _Float16* kg16   = (_Float16*)alloc((size_t)S_ * D_ * 2);
    _Float16* vg16   = (_Float16*)alloc((size_t)S_ * D_ * 2);
    _Float16* ff16   = (_Float16*)alloc((size_t)S_ * FF_ * 2);
    _Float16* Wt     = (_Float16*)alloc((size_t)D_ * V_ * 2);   // largest weight

    auto cvt = [&](const float* W, int K, int N) {
        long long tot = (long long)K * N;
        k_cvtT<<<dim3((unsigned)((tot + 255) / 256)), 256, 0, stream>>>(W, Wt, K, N);
    };
    auto gemm = [&](const _Float16* A, const float* bias, const float* resid,
                    float* C32, _Float16* C16, int M, int N, int K, float scale, int act) {
        k_gemm<<<dim3(N / 64, M / 128), 128, 0, stream>>>(A, Wt, bias, resid, C32, C16, M, N, K, scale, act);
    };

    const float qscale = 0.125f;   // 1/sqrt(64)

    // embeddings + LN
    k_embed<<<dim3(S_ * D_ / 256), 256, 0, stream>>>(x, we, pe, te, tmp32);
    k_layernorm<<<dim3(S_), 128, 0, stream>>>(tmp32, elng, elnb, h32, h16);

    for (int l = 0; l < 4; ++l) {
        int b = 8 + l * 22;
        const float* Wq  = (const float*)d_in[b + 0];  const float* bq  = (const float*)d_in[b + 1];
        const float* Wk  = (const float*)d_in[b + 2];  const float* bk  = (const float*)d_in[b + 3];
        const float* Wv  = (const float*)d_in[b + 4];  const float* bv  = (const float*)d_in[b + 5];
        const float* Wqg = (const float*)d_in[b + 6];  const float* bqg = (const float*)d_in[b + 7];
        const float* Wkg = (const float*)d_in[b + 8];  const float* bkg = (const float*)d_in[b + 9];
        const float* Wvg = (const float*)d_in[b + 10]; const float* bvg = (const float*)d_in[b + 11];
        const float* Wo  = (const float*)d_in[b + 12]; const float* bo  = (const float*)d_in[b + 13];
        const float* l1g = (const float*)d_in[b + 14]; const float* l1b = (const float*)d_in[b + 15];
        const float* Wi  = (const float*)d_in[b + 16]; const float* bi  = (const float*)d_in[b + 17];
        const float* Wf  = (const float*)d_in[b + 18]; const float* bf  = (const float*)d_in[b + 19];
        const float* l2g = (const float*)d_in[b + 20]; const float* l2b = (const float*)d_in[b + 21];

        cvt(Wq, D_, D_); gemm(h16, bq, nullptr, nullptr, q16, S_, D_, D_, qscale, 0);
        cvt(Wk, D_, D_); gemm(h16, bk, nullptr, nullptr, k16, S_, D_, D_, 1.0f, 0);
        cvt(Wv, D_, D_); gemm(h16, bv, nullptr, nullptr, v16, S_, D_, D_, 1.0f, 0);
        k_vtrans<<<dim3(S_ * D_ / 256), 256, 0, stream>>>(v16, vT16);

        k_local_attn<<<dim3(WW_ / 16, NC_, H_), 32, 0, stream>>>(q16, k16, vT16, attn16, amask, gmask);

        cvt(Wqg, D_, D_); gemm(h16, bqg, nullptr, nullptr, qg16, S_, D_, D_, qscale, 0);
        cvt(Wkg, D_, D_); gemm(h16, bkg, nullptr, nullptr, kg16, S_, D_, D_, 1.0f, 0);
        cvt(Wvg, D_, D_); gemm(h16, bvg, nullptr, nullptr, vg16, S_, D_, D_, 1.0f, 0);
        k_global_attn<<<dim3(H_ * G_), 128, 0, stream>>>(qg16, kg16, vg16, attn16, amask);

        cvt(Wo, D_, D_); gemm(attn16, bo, h32, tmp32, nullptr, S_, D_, D_, 1.0f, 0);
        k_layernorm<<<dim3(S_), 128, 0, stream>>>(tmp32, l1g, l1b, h32, h16);

        cvt(Wi, D_, FF_); gemm(h16, bi, nullptr, nullptr, ff16, S_, FF_, D_, 1.0f, 1);
        cvt(Wf, FF_, D_); gemm(ff16, bf, h32, tmp32, nullptr, S_, D_, FF_, 1.0f, 0);
        k_layernorm<<<dim3(S_), 128, 0, stream>>>(tmp32, l2g, l2b, h32, h16);
    }

    // logits = h @ W_out + b_out  (straight to d_out), then append h
    cvt(W_out, D_, V_);
    gemm(h16, b_out, nullptr, out, nullptr, S_, V_, D_, 1.0f, 0);
    k_copyf<<<dim3(S_ * D_ / 256), 256, 0, stream>>>(h32, out + (size_t)S_ * V_, S_ * D_);
}
